// RecurrentNEFLayer_26216480374911
// MI455X (gfx1250) — matrix operand order repaired and checked by reading, not compiled, not run on hardware
//
#include <hip/hip_runtime.h>
#include <hip/hip_bf16.h>

// ---------------------------------------------------------------------------
// RecurrentNEFLayer on MI455X (gfx1250, wave32, WMMA f16 16x16x32).
//
// Per workgroup: one 16-row batch tile. State s and activations a live in LDS
// across all 512 timesteps. Weights pre-converted to f16 in workspace:
//   Ef16  [2048 x 256]  = gain[n] * encoders[n][k]      (gain folded in)
//   SDT   [ 128 x 2048] = state_decoders^T
//   DecT  [  64 x 2048] = decoders^T
// GEMM1: a(16x2048) = relu(Xaug(16x256) @ Ef16^T + bias)   -> 128 WMMA / wave
// GEMM2: s(16x128)  = a @ SD                                ->  64 WMMA / wave
// Epilogue: out(16x64) = a_final @ Dec                      ->  64 WMMA / wave
//
// v3: anti-LICM via an opaque zero SGPR *offset* (not an opaque pointer).
// v2's "+v" pointer barrier stopped the scratch spills but broke address-space
// inference, so weights were fetched with flat_load_b128 (couples LOADcnt AND
// DScnt -> s_wait_loadcnt_dscnt before every WMMA). Adding an opaque zoff to a
// kernarg-rooted pointer keeps the loads un-hoistable while letting the
// compiler prove global address space -> global_load_b128 with SGPR base and
// loadcnt-only waits, decoupled from the LDS pipeline.
// ---------------------------------------------------------------------------

typedef _Float16 half8  __attribute__((ext_vector_type(8)));
typedef _Float16 half16 __attribute__((ext_vector_type(16)));
typedef float    float8 __attribute__((ext_vector_type(8)));
typedef float    floatv4 __attribute__((ext_vector_type(4)));

#define B_SZ   256
#define T_SZ   512
#define DIN    128
#define DST    128
#define DAUG   256   // DIN + DST
#define NN     2048
#define DOUT   64

#define XA_STRIDE 264   // DAUG + 8 halfs pad (row stride 528B, 16B-aligned)
#define A_STRIDE  2056  // NN + 8 halfs pad  (row stride 4112B, 16B-aligned)

static __device__ __forceinline__ half16 cat8(half8 lo, half8 hi) {
  return __builtin_shufflevector(lo, hi, 0,1,2,3,4,5,6,7,8,9,10,11,12,13,14,15);
}

static __device__ __forceinline__ half8 cvt8(floatv4 a, floatv4 b) {
  half8 h;
#pragma unroll
  for (int i = 0; i < 4; ++i) { h[i] = (_Float16)a[i]; h[i+4] = (_Float16)b[i]; }
  return h;
}

// ---- prep kernels: fold gain into encoders; transpose + cvt decoders ------
__global__ void prep_encoders_kernel(const float* __restrict__ enc,
                                     const float* __restrict__ gain,
                                     _Float16* __restrict__ Ef16) {
  int i = blockIdx.x * blockDim.x + threadIdx.x;
  if (i < NN * DAUG) {
    int n = i / DAUG;
    Ef16[i] = (_Float16)(gain[n] * enc[i]);
  }
}

__global__ void prep_transpose_kernel(const float* __restrict__ src,
                                      _Float16* __restrict__ dst,
                                      int rows, int cols) {
  // src: rows x cols row-major -> dst: cols x rows row-major (f16)
  int i = blockIdx.x * blockDim.x + threadIdx.x;
  if (i < rows * cols) {
    int r = i / cols, c = i - r * cols;
    dst[(size_t)c * rows + r] = (_Float16)src[i];
  }
}

// ---- main recurrent kernel -------------------------------------------------
__global__ __launch_bounds__(256) void nef_recurrent_kernel(
    const float* __restrict__ seq,       // (B, T, DIN) f32
    const float* __restrict__ bias,      // (NN) f32
    const _Float16* __restrict__ Ef16,   // (NN, DAUG)
    const _Float16* __restrict__ SDT,    // (DST, NN)
    const _Float16* __restrict__ DecT,   // (DOUT, NN)
    float* __restrict__ out)             // (B, DOUT) f32
{
  __shared__ _Float16 xaug[16 * XA_STRIDE]; // [u_t | s] f16, 16 rows
  __shared__ _Float16 alds[16 * A_STRIDE];  // a (16 x 2048) f16

  const int tid  = threadIdx.x;
  const int lane = tid & 31;
  const int wave = tid >> 5;        // 0..7
  const int wg   = blockIdx.x;      // batch tile 0..15
  const int hi   = lane >> 4;       // lane half
  const int l15  = lane & 15;
  const int aoff = hi ? 8 : 0;      // A-fragment K sub-offset
  const int boff = hi ? 8 : 0;      // B-fragment K sub-offset

  // cooperative u-loader mapping: each thread owns 8 f32 of one Xaug row
  const int urow = tid >> 4;        // 0..15
  const int ucol = (tid & 15) * 8;  // 0..120

  // ---- init: load u_0, zero the state half ----
  {
    const float* src = seq + ((size_t)(wg * 16 + urow) * T_SZ) * DIN + ucol;
    floatv4 f0 = *(const floatv4*)(src);
    floatv4 f1 = *(const floatv4*)(src + 4);
    *(half8*)(&xaug[urow * XA_STRIDE + ucol]) = cvt8(f0, f1);
    half8 z = {};
    *(half8*)(&xaug[urow * XA_STRIDE + DIN + ucol]) = z;
  }

  // ---- deliberately hoist: per-lane bias scalars for this wave's 16 N-tiles
  float breg[16];
#pragma unroll
  for (int nt = 0; nt < 16; ++nt)
    breg[nt] = bias[wave * 256 + nt * 16 + l15];

  __syncthreads();

  // per-lane weight base pointers (still rooted at kernel args -> global AS)
  const _Float16* ebase = Ef16 + (size_t)(wave * 256 + l15) * DAUG;  // + nt*16*DAUG
  const _Float16* sbase = SDT  + (size_t)(wave * 16  + l15) * NN;

  for (int t = 0; t < T_SZ; ++t) {
    // Opaque zero offset in an SGPR: the weight-load addresses depend on it,
    // so LICM cannot hoist the (t-invariant) loads out of the 512-step loop,
    // yet the pointers remain kernarg-rooted so the backend still proves
    // global address space (global_load_b128, loadcnt-only waits).
    size_t zoff = 0;
    asm volatile("" : "+s"(zoff));
    const _Float16* ept = ebase + zoff;
    const _Float16* spt = sbase + zoff;

    // ===== GEMM1: a = relu(Xaug @ E^T + bias), wave w -> neurons [256w,256w+256)
    half16 afrag[8];
#pragma unroll
    for (int kt = 0; kt < 8; ++kt) {
      half8 lo = *(const half8*)(&xaug[l15 * XA_STRIDE + kt * 32 + aoff]);
      half8 hb = *(const half8*)(&xaug[l15 * XA_STRIDE + kt * 32 + 16 + aoff]);
      afrag[kt] = cat8(lo, hb);
    }
#pragma unroll 2
    for (int nt = 0; nt < 16; ++nt) {
      const _Float16* ep = ept + (size_t)nt * 16 * DAUG;  // lane's B column row
      float8 c = {};
#pragma unroll
      for (int kt = 0; kt < 8; ++kt) {
        half8 blo = *(const half8*)(ep + kt * 32 + boff);
        half8 bhi = *(const half8*)(ep + kt * 32 + boff + 16);
        c = __builtin_amdgcn_wmma_f32_16x16x32_f16(
                false, afrag[kt], false, cat8(blo, bhi),
                (short)0, c, false, false);
      }
      const float bn = breg[nt];
      const int ncol = wave * 256 + nt * 16 + l15;
#pragma unroll
      for (int v = 0; v < 8; ++v) {
        float r = c[v] + bn;
        r = r > 0.f ? r : 0.f;
        alds[(v + 8 * hi) * A_STRIDE + ncol] = (_Float16)r;
      }
    }
    __syncthreads();

    if (t + 1 < T_SZ) {
      // ===== GEMM2: s = a @ SD, wave w -> state cols [16w, 16w+16)
      const int j = wave * 16 + l15;
      float8 c = {};
#pragma unroll 4
      for (int kt = 0; kt < 64; ++kt) {
        half8 alo = *(const half8*)(&alds[l15 * A_STRIDE + kt * 32 + aoff]);
        half8 ahi = *(const half8*)(&alds[l15 * A_STRIDE + kt * 32 + 16 + aoff]);
        half8 blo = *(const half8*)(spt + kt * 32 + boff);
        half8 bhi = *(const half8*)(spt + kt * 32 + boff + 16);
        c = __builtin_amdgcn_wmma_f32_16x16x32_f16(
                false, cat8(alo, ahi), false, cat8(blo, bhi),
                (short)0, c, false, false);
      }
#pragma unroll
      for (int v = 0; v < 8; ++v)
        xaug[(v + 8 * hi) * XA_STRIDE + DIN + j] = (_Float16)c[v];

      // ---- stage u_{t+1} into the input half of Xaug ----
      const float* src = seq + ((size_t)(wg * 16 + urow) * T_SZ + (t + 1)) * DIN + ucol;
      floatv4 f0 = *(const floatv4*)(src);
      floatv4 f1 = *(const floatv4*)(src + 4);
      *(half8*)(&xaug[urow * XA_STRIDE + ucol]) = cvt8(f0, f1);
      __syncthreads();
    }
  }

  // ===== epilogue: out = a_final @ Dec  (waves 0..3, 16x16 tiles over DOUT=64)
  if (wave < 4) {
    const int o = wave * 16 + l15;
    const _Float16* dp = DecT + (size_t)o * NN;
    float8 c = {};
#pragma unroll 4
    for (int kt = 0; kt < 64; ++kt) {
      half8 alo = *(const half8*)(&alds[l15 * A_STRIDE + kt * 32 + aoff]);
      half8 ahi = *(const half8*)(&alds[l15 * A_STRIDE + kt * 32 + 16 + aoff]);
      half8 blo = *(const half8*)(dp + kt * 32 + boff);
      half8 bhi = *(const half8*)(dp + kt * 32 + boff + 16);
      c = __builtin_amdgcn_wmma_f32_16x16x32_f16(
              false, cat8(alo, ahi), false, cat8(blo, bhi),
              (short)0, c, false, false);
    }
#pragma unroll
    for (int v = 0; v < 8; ++v)
      out[(size_t)(wg * 16 + v + 8 * hi) * DOUT + o] = c[v];
  }
}

extern "C" void kernel_launch(void* const* d_in, const int* in_sizes, int n_in,
                              void* d_out, int out_size, void* d_ws, size_t ws_size,
                              hipStream_t stream) {
  const float* seq  = (const float*)d_in[0];
  const float* enc  = (const float*)d_in[1];
  const float* gain = (const float*)d_in[2];
  const float* bias = (const float*)d_in[3];
  const float* sd   = (const float*)d_in[4];
  const float* dec  = (const float*)d_in[5];
  float* out = (float*)d_out;

  _Float16* Ef16 = (_Float16*)d_ws;                 // 2048*256  f16 = 1   MB
  _Float16* SDT  = Ef16 + (size_t)NN * DAUG;        //  128*2048 f16 = 512 KB
  _Float16* DecT = SDT  + (size_t)DST * NN;         //   64*2048 f16 = 256 KB

  prep_encoders_kernel<<<(NN * DAUG + 255) / 256, 256, 0, stream>>>(enc, gain, Ef16);
  prep_transpose_kernel<<<(NN * DST + 255) / 256, 256, 0, stream>>>(sd, SDT, NN, DST);
  prep_transpose_kernel<<<(NN * DOUT + 255) / 256, 256, 0, stream>>>(dec, DecT, NN, DOUT);

  nef_recurrent_kernel<<<B_SZ / 16, 256, 0, stream>>>(seq, bias, Ef16, SDT, DecT, out);
}
